// ASGAT3_54176717471755
// MI455X (gfx1250) — matrix operand order repaired and checked by reading, not compiled, not run on hardware
//
#include <hip/hip_runtime.h>
#include <hip/hip_bf16.h>

// ---------------------------------------------------------------------------
// ASGAT forward loss on MI455X (gfx1250).
// Pipeline:
//   0) embedding gather -> f16 [B*L, 320]   (E=300 padded to 320 for K%32==0)
//   1) weight converts  -> f16 (W_ih padded, W_hh, gat_W head0 transposed)
//   2) Z_{f,b} = X @ W_ih^T      (WMMA GEMM, 1x4 tile register blocking)
//   3) LSTM recurrence (grid=2: fwd/bwd). h in LDS, c in wave accumulators,
//      each wave owns all 4 gate tiles for its (m,j) slice -> in-lane fusion.
//   4) h0 = text_out @ gat_W0    (WMMA GEMM, stores transposed f16 hT[b][n][l])
//   5) e1/e2 vectors, masked-softmax attention rows -> attn f16
//   6) C2 = attn @ h0            (WMMA GEMM per batch)
//   7) double-ELU span mean-pool (head-0 block only), FC + CE + mean
// ---------------------------------------------------------------------------

#define Bn  32
#define Ln  256
#define En  300
#define EP  320
#define Hn  256
#define F2n 512
#define An  2
#define Pn  3

typedef __attribute__((ext_vector_type(16))) _Float16 v16h;
typedef __attribute__((ext_vector_type(8)))  _Float16 v8h;
typedef __attribute__((ext_vector_type(8)))  float    v8f;

// ---- WMMA helpers ----------------------------------------------------------
// A (16x32 f16): lane<16 holds row m0+lane, K k0..k0+15;
//                lane>=16 holds row m0+lane-16, K k0+16..k0+31.
__device__ __forceinline__ v16h ldA(const _Float16* A, int lda, int m0, int k0, int lane) {
  const int l  = lane & 15;
  const int ks = (lane >> 4) << 4;                 // 0 or 16
  const _Float16* p = A + (size_t)(m0 + l) * lda + (k0 + ks);
  v8h lo = *(const v8h*)p;                          // K = k0+ks .. +7
  v8h hi = *(const v8h*)(p + 8);                    // K = k0+ks+8 .. +15
  v16h r;
#pragma unroll
  for (int i = 0; i < 8; ++i) { r[i] = lo[i]; r[i + 8] = hi[i]; }
  return r;
}

// B (32x16 f16) from K-contiguous storage Bt[n][k] (row-major, ldb halves):
// lane<16 holds col n0+lane, K k0..k0+15; lane>=16 same col, K k0+16..k0+31.
__device__ __forceinline__ v16h ldB(const _Float16* Bt, int ldb, int n0, int k0, int lane) {
  const int n  = n0 + (lane & 15);
  const int ko = (lane >> 4) << 4;                 // 0 or 16
  return *(const v16h*)(Bt + (size_t)n * ldb + (k0 + ko));
}

__device__ __forceinline__ v8f wmma32(v16h a, v16h b, v8f c) {
  return __builtin_amdgcn_wmma_f32_16x16x32_f16(false, a, false, b, (short)0, c, false, false);
}

__device__ __forceinline__ float sigm(float x) { return 1.f / (1.f + __expf(-x)); }
__device__ __forceinline__ float elu2(float v) {
  float e = v > 0.f ? v : (__expf(v) - 1.f);
  return e > 0.f ? e : (__expf(e) - 1.f);
}

// ---- 0) embedding gather + f16 convert (pad E 300->320) --------------------
__global__ void k_embed(const int* __restrict__ tix, const float* __restrict__ emb,
                        _Float16* __restrict__ xf16) {
  int i = blockIdx.x * 256 + threadIdx.x;
  if (i >= Bn * Ln * EP) return;
  int bl = i / EP, e = i - bl * EP;
  xf16[i] = (e < En) ? (_Float16)emb[(size_t)tix[bl] * En + e] : (_Float16)0.f;
}

// ---- 1) weight conversion kernels ------------------------------------------
__global__ void k_conv_wih(const float* __restrict__ w, _Float16* __restrict__ o) {
  int i = blockIdx.x * 256 + threadIdx.x;
  if (i >= 4 * Hn * EP) return;
  int n = i / EP, k = i - n * EP;
  o[i] = (k < En) ? (_Float16)w[(size_t)n * En + k] : (_Float16)0.f;
}
__global__ void k_conv_whh(const float* __restrict__ w, _Float16* __restrict__ o) {
  int i = blockIdx.x * 256 + threadIdx.x;
  if (i < 4 * Hn * Hn) o[i] = (_Float16)w[i];
}
__global__ void k_conv_gw0t(const float* __restrict__ W, _Float16* __restrict__ o) {
  int i = blockIdx.x * 256 + threadIdx.x;
  if (i >= F2n * F2n) return;
  int n = i / F2n, k = i - n * F2n;
  o[i] = (_Float16)W[(size_t)k * F2n + n];          // transpose: o[n][k] = W[k][n]
}

// ---- 2) WMMA GEMM, 1x4 register blocking: C[M,N] = A[M,K] * Bt[N,K]^T ------
// One wave computes a 16x64 strip: A fragment re-used across 4 WMMAs.
__global__ __launch_bounds__(256) void gemm_xw(const _Float16* __restrict__ A,
                                               const _Float16* __restrict__ Bt,
                                               float* __restrict__ C,
                                               int M, int N, int K, int lda, int ldb) {
  int gw = blockIdx.x * 8 + (threadIdx.x >> 5);
  int lane = threadIdx.x & 31;
  int nquads = N >> 6;                               // 64-wide N strips
  int tm = (gw / nquads) << 4;
  int tn = (gw - (gw / nquads) * nquads) << 6;
  if (tm >= M) return;
  v8f acc0 = {}, acc1 = {}, acc2 = {}, acc3 = {};
  for (int k0 = 0; k0 < K; k0 += 32) {
    if (k0 + 32 < K) __builtin_prefetch(Bt + (size_t)(tn + (lane & 15)) * ldb + k0 + 32, 0, 1);
    v16h a = ldA(A, lda, tm, k0, lane);              // 1 A fragment : 4 WMMAs
    acc0 = wmma32(a, ldB(Bt, ldb, tn,      k0, lane), acc0);
    acc1 = wmma32(a, ldB(Bt, ldb, tn + 16, k0, lane), acc1);
    acc2 = wmma32(a, ldB(Bt, ldb, tn + 32, k0, lane), acc2);
    acc3 = wmma32(a, ldB(Bt, ldb, tn + 48, k0, lane), acc3);
  }
  int row0 = tm + ((lane >> 4) << 3);
  int col  = tn + (lane & 15);
#pragma unroll
  for (int r = 0; r < 8; ++r) {
    float* cr = C + (size_t)(row0 + r) * N + col;
    cr[0] = acc0[r]; cr[16] = acc1[r]; cr[32] = acc2[r]; cr[48] = acc3[r];
  }
}

// ---- 3) BiLSTM recurrence: grid=2 (dir), block=1024 (32 waves) -------------
__global__ __launch_bounds__(1024) void lstm_rec(const float* __restrict__ Zf,
                                                 const float* __restrict__ Zb,
                                                 const _Float16* __restrict__ Whhf,
                                                 const _Float16* __restrict__ Whhb,
                                                 const float* __restrict__ bf,
                                                 const float* __restrict__ bb,
                                                 _Float16* __restrict__ tout) {
  const int dir = blockIdx.x;
  const float*    Z    = dir ? Zb   : Zf;
  const _Float16* Whh  = dir ? Whhb : Whhf;
  const float*    bias = dir ? bb   : bf;

  __shared__ _Float16 h_s[Bn * Hn];                 // 16 KB: h_{t-1}, f16
  const int tid = threadIdx.x, lane = tid & 31, wave = tid >> 5;
  for (int i = tid; i < Bn * Hn; i += 1024) h_s[i] = (_Float16)0.f;

  const int m0   = (wave & 1) << 4;                 // m-tile: batch rows
  const int jt   = wave >> 1;                       // 0..15: H column tile
  const int jcol = (jt << 4) + (lane & 15);
  const float bi = bias[jcol],        bff = bias[Hn + jcol];
  const float bg = bias[2*Hn + jcol], bo  = bias[3*Hn + jcol];
  v8f c_acc = {};                                   // cell state lives in regs
  __syncthreads();

  for (int s = 0; s < Ln; ++s) {
    const int t = dir ? (Ln - 1 - s) : s;
    v8f ai = {}, af = {}, ag = {}, ao = {};
#pragma unroll
    for (int k0 = 0; k0 < Hn; k0 += 32) {
      v16h a = ldA(h_s, Hn, m0, k0, lane);          // shared per 4 gate tiles
      ai = wmma32(a, ldB(Whh, Hn, 0 * Hn + (jt << 4), k0, lane), ai);
      af = wmma32(a, ldB(Whh, Hn, 1 * Hn + (jt << 4), k0, lane), af);
      ag = wmma32(a, ldB(Whh, Hn, 2 * Hn + (jt << 4), k0, lane), ag);
      ao = wmma32(a, ldB(Whh, Hn, 3 * Hn + (jt << 4), k0, lane), ao);
    }
    __syncthreads();                                // all reads of h_s done
#pragma unroll
    for (int r = 0; r < 8; ++r) {
      const int row = m0 + ((lane >> 4) << 3) + r;  // batch index
      const float* zr = Z + ((size_t)(row * Ln + t)) * (4 * Hn);
      float zi = ai[r] + zr[jcol]          + bi;
      float zf = af[r] + zr[Hn + jcol]     + bff;
      float zg = ag[r] + zr[2*Hn + jcol]   + bg;
      float zo = ao[r] + zr[3*Hn + jcol]   + bo;
      float cv = sigm(zf) * c_acc[r] + sigm(zi) * tanhf(zg);
      c_acc[r] = cv;
      float hv = sigm(zo) * tanhf(cv);
      h_s[row * Hn + jcol] = (_Float16)hv;
      tout[((size_t)(row * Ln + t)) * F2n + dir * Hn + jcol] = (_Float16)hv;
    }
    __syncthreads();                                // h_t visible to all waves
  }
}

// ---- 4) GAT head-0 features: hT[b][n][l] = (text_out @ W0)(b,l,n), f16 -----
__global__ __launch_bounds__(256) void gemm_feat(const _Float16* __restrict__ A,
                                                 const _Float16* __restrict__ Bt,
                                                 _Float16* __restrict__ hT) {
  int gw = blockIdx.x * 8 + (threadIdx.x >> 5);
  int lane = threadIdx.x & 31;
  const int nquads = F2n >> 6;                      // 8
  int tm = (gw / nquads) << 4;
  int tn = (gw - (gw / nquads) * nquads) << 6;
  v8f acc0 = {}, acc1 = {}, acc2 = {}, acc3 = {};
  for (int k0 = 0; k0 < F2n; k0 += 32) {
    if (k0 + 32 < F2n) __builtin_prefetch(Bt + (size_t)(tn + (lane & 15)) * F2n + k0 + 32, 0, 1);
    v16h a = ldA(A, F2n, tm, k0, lane);
    acc0 = wmma32(a, ldB(Bt, F2n, tn,      k0, lane), acc0);
    acc1 = wmma32(a, ldB(Bt, F2n, tn + 16, k0, lane), acc1);
    acc2 = wmma32(a, ldB(Bt, F2n, tn + 32, k0, lane), acc2);
    acc3 = wmma32(a, ldB(Bt, F2n, tn + 48, k0, lane), acc3);
  }
  int row0 = tm + ((lane >> 4) << 3);
  int col  = tn + (lane & 15);
#pragma unroll
  for (int r = 0; r < 8; ++r) {
    int row = row0 + r, b = row >> 8, l = row & 255;
    _Float16* hb = hT + (size_t)b * F2n * Ln + l;
    hb[(size_t)(col     ) * Ln] = (_Float16)acc0[r];
    hb[(size_t)(col + 16) * Ln] = (_Float16)acc1[r];
    hb[(size_t)(col + 32) * Ln] = (_Float16)acc2[r];
    hb[(size_t)(col + 48) * Ln] = (_Float16)acc3[r];
  }
}

// ---- 5a) e1/e2 vectors -----------------------------------------------------
__global__ void k_e1e2(const _Float16* __restrict__ hT, const float* __restrict__ a,
                       float* __restrict__ e1, float* __restrict__ e2) {
  int b = blockIdx.x, l = threadIdx.x;
  const _Float16* h = hT + (size_t)b * F2n * Ln;
  float s1 = 0.f, s2 = 0.f;
  for (int n = 0; n < F2n; ++n) {
    float hv = (float)h[n * Ln + l];
    s1 += hv * a[n];
    s2 += hv * a[F2n + n];
  }
  e1[b * Ln + l] = s1;
  e2[b * Ln + l] = s2;
}

// ---- 5b) masked softmax row -> attn f16 ------------------------------------
__global__ void k_attn(const float* __restrict__ e1, const float* __restrict__ e2,
                       const float* __restrict__ adj, _Float16* __restrict__ attn) {
  int bi = blockIdx.x, b = bi >> 8, i = bi & 255, j = threadIdx.x;
  float v = e1[b * Ln + i] + e2[b * Ln + j];
  v = v > 0.f ? v : 0.2f * v;                       // LeakyReLU alpha=0.2
  float m = adj[((size_t)b * Ln + i) * Ln + j];
  float val = m > 0.f ? v : -9e15f;
  __shared__ float red[Ln];
  red[j] = val; __syncthreads();
  for (int s = 128; s > 0; s >>= 1) { if (j < s) red[j] = fmaxf(red[j], red[j + s]); __syncthreads(); }
  float mx = red[0]; __syncthreads();
  float ex = __expf(val - mx);
  red[j] = ex; __syncthreads();
  for (int s = 128; s > 0; s >>= 1) { if (j < s) red[j] += red[j + s]; __syncthreads(); }
  attn[((size_t)b * Ln + i) * Ln + j] = (_Float16)(ex / red[0]);
}

// ---- 6) C2[b] = attn[b] @ h0[b] (via hT), 1x4 blocking ---------------------
__global__ __launch_bounds__(256) void gemm_attnh(const _Float16* __restrict__ attn,
                                                  const _Float16* __restrict__ hT,
                                                  float* __restrict__ C2) {
  int gw = blockIdx.x * 8 + (threadIdx.x >> 5);
  int lane = threadIdx.x & 31;
  const int quadsPerB = (Ln >> 4) * (F2n >> 6);     // 16 * 8 = 128
  int b   = gw / quadsPerB;
  int rem = gw - b * quadsPerB;
  int tm = (rem / (F2n >> 6)) << 4;
  int tn = (rem - (rem / (F2n >> 6)) * (F2n >> 6)) << 6;
  const _Float16* A  = attn + (size_t)b * Ln * Ln;
  const _Float16* Bt = hT   + (size_t)b * F2n * Ln;
  v8f acc0 = {}, acc1 = {}, acc2 = {}, acc3 = {};
#pragma unroll
  for (int k0 = 0; k0 < Ln; k0 += 32) {
    v16h a = ldA(A, Ln, tm, k0, lane);
    acc0 = wmma32(a, ldB(Bt, Ln, tn,      k0, lane), acc0);
    acc1 = wmma32(a, ldB(Bt, Ln, tn + 16, k0, lane), acc1);
    acc2 = wmma32(a, ldB(Bt, Ln, tn + 32, k0, lane), acc2);
    acc3 = wmma32(a, ldB(Bt, Ln, tn + 48, k0, lane), acc3);
  }
  int row0 = tm + ((lane >> 4) << 3);
  int col  = tn + (lane & 15);
  float* C = C2 + (size_t)b * Ln * F2n;
#pragma unroll
  for (int r = 0; r < 8; ++r) {
    float* cr = C + (size_t)(row0 + r) * F2n + col;
    cr[0] = acc0[r]; cr[16] = acc1[r]; cr[32] = acc2[r]; cr[48] = acc3[r];
  }
}

// ---- 7a) double-ELU span mean-pool (head-0 block only) ---------------------
__global__ void k_pool(const float* __restrict__ C2, const int* __restrict__ st,
                       const int* __restrict__ en, float* __restrict__ asp) {
  int ba = blockIdx.x, b = ba >> 1;
  int s0 = st[ba], s1 = en[ba];
  int f = threadIdx.x;
  float s = 0.f;
  for (int l = s0; l < s1; ++l) s += elu2(C2[((size_t)b * Ln + l) * F2n + f]);
  asp[(size_t)ba * F2n + f] = s / (float)(s1 - s0);
}

// ---- 7b) FC + log-softmax + CE + mean --------------------------------------
__global__ void k_final(const float* __restrict__ asp, const float* __restrict__ fcW,
                        const float* __restrict__ fcb, const int* __restrict__ pol,
                        float* __restrict__ out) {
  __shared__ float ce_s[Bn * An];
  int t = threadIdx.x;
  if (t < Bn * An) {
    const float* av = asp + (size_t)t * F2n;
    float l0 = fcb[0], l1 = fcb[1], l2 = fcb[2];
    for (int f = 0; f < F2n; ++f) {
      float v = av[f];
      l0 += v * fcW[f]; l1 += v * fcW[F2n + f]; l2 += v * fcW[2 * F2n + f];
    }
    float mx  = fmaxf(l0, fmaxf(l1, l2));
    float lse = mx + __logf(__expf(l0 - mx) + __expf(l1 - mx) + __expf(l2 - mx));
    int p = pol[t];
    float lp = (p == 0 ? l0 : (p == 1 ? l1 : l2)) - lse;
    ce_s[t] = -lp;
  }
  __syncthreads();
  if (t == 0) {
    float s = 0.f;
    for (int i = 0; i < Bn * An; ++i) s += ce_s[i];
    out[0] = s / (float)(Bn * An);
  }
}

// ---------------------------------------------------------------------------
extern "C" void kernel_launch(void* const* d_in, const int* in_sizes, int n_in,
                              void* d_out, int out_size, void* d_ws, size_t ws_size,
                              hipStream_t stream) {
  const int*   tix   = (const int*)  d_in[0];
  const int*   astar = (const int*)  d_in[1];
  const int*   aend  = (const int*)  d_in[2];
  const int*   pol   = (const int*)  d_in[3];
  const float* adj   = (const float*)d_in[4];
  const float* emb   = (const float*)d_in[5];
  const float* wihf  = (const float*)d_in[6];
  const float* whhf  = (const float*)d_in[7];
  const float* bf    = (const float*)d_in[8];
  const float* wihb  = (const float*)d_in[9];
  const float* whhb  = (const float*)d_in[10];
  const float* bb    = (const float*)d_in[11];
  const float* gatW  = (const float*)d_in[12];
  const float* gata  = (const float*)d_in[13];
  const float* fcW   = (const float*)d_in[14];
  const float* fcb   = (const float*)d_in[15];
  float* out = (float*)d_out;

  // workspace carve (~113 MB)
  char* w = (char*)d_ws;
  size_t o = 0;
  auto carve = [&](size_t bytes) { char* p = w + o; o = (o + bytes + 255) & ~(size_t)255; return p; };
  _Float16* xf16   = (_Float16*)carve((size_t)Bn * Ln * EP * 2);       // 5.24 MB
  _Float16* wfF    = (_Float16*)carve((size_t)4 * Hn * EP * 2);
  _Float16* wfB    = (_Float16*)carve((size_t)4 * Hn * EP * 2);
  _Float16* whF    = (_Float16*)carve((size_t)4 * Hn * Hn * 2);
  _Float16* whB    = (_Float16*)carve((size_t)4 * Hn * Hn * 2);
  _Float16* g0t    = (_Float16*)carve((size_t)F2n * F2n * 2);
  float*    Zf     = (float*)   carve((size_t)Bn * Ln * 4 * Hn * 4);   // 33.5 MB
  float*    Zb     = (float*)   carve((size_t)Bn * Ln * 4 * Hn * 4);   // 33.5 MB
  _Float16* tout   = (_Float16*)carve((size_t)Bn * Ln * F2n * 2);      // 8.4 MB
  _Float16* hT     = (_Float16*)carve((size_t)Bn * F2n * Ln * 2);      // 8.4 MB
  float*    e1     = (float*)   carve((size_t)Bn * Ln * 4);
  float*    e2     = (float*)   carve((size_t)Bn * Ln * 4);
  _Float16* attn   = (_Float16*)carve((size_t)Bn * Ln * Ln * 2);       // 4.2 MB
  float*    C2     = (float*)   carve((size_t)Bn * Ln * F2n * 4);      // 16.8 MB
  float*    asp    = (float*)   carve((size_t)Bn * An * F2n * 4);
  (void)ws_size; (void)n_in; (void)in_sizes; (void)out_size;

  // 0) gather + converts
  k_embed   <<<(Bn * Ln * EP + 255) / 256, 256, 0, stream>>>(tix, emb, xf16);
  k_conv_wih<<<(4 * Hn * EP + 255) / 256, 256, 0, stream>>>(wihf, wfF);
  k_conv_wih<<<(4 * Hn * EP + 255) / 256, 256, 0, stream>>>(wihb, wfB);
  k_conv_whh<<<(4 * Hn * Hn + 255) / 256, 256, 0, stream>>>(whhf, whF);
  k_conv_whh<<<(4 * Hn * Hn + 255) / 256, 256, 0, stream>>>(whhb, whB);
  k_conv_gw0t<<<(F2n * F2n + 255) / 256, 256, 0, stream>>>(gatW, g0t);

  // 2) hoisted input projections (WMMA, 16x64 strips per wave)
  {
    const int waves = (Bn * Ln / 16) * (4 * Hn / 64);                  // 8192
    gemm_xw<<<waves / 8, 256, 0, stream>>>(xf16, wfF, Zf, Bn * Ln, 4 * Hn, EP, EP, EP);
    gemm_xw<<<waves / 8, 256, 0, stream>>>(xf16, wfB, Zb, Bn * Ln, 4 * Hn, EP, EP, EP);
  }

  // 3) BiLSTM recurrence (fwd + bwd concurrently)
  lstm_rec<<<2, 1024, 0, stream>>>(Zf, Zb, whF, whB, bf, bb, tout);

  // 4) GAT head-0 feature transform (WMMA, transposed f16 output)
  gemm_feat<<<((Bn * Ln / 16) * (F2n / 64)) / 8, 256, 0, stream>>>(tout, g0t, hT);

  // 5) attention scores + masked softmax
  k_e1e2<<<Bn, Ln, 0, stream>>>(hT, gata, e1, e2);
  k_attn<<<Bn * Ln, Ln, 0, stream>>>(e1, e2, adj, attn);

  // 6) attn @ h (WMMA, per batch)
  gemm_attnh<<<(Bn * (Ln / 16) * (F2n / 64)) / 8, 256, 0, stream>>>(attn, hT, C2);

  // 7) pool + FC + CE
  k_pool <<<Bn * An, F2n, 0, stream>>>(C2, astar, aend, asp);
  k_final<<<1, 64, 0, stream>>>(asp, fcW, fcb, pol, out);
}
